// svgp_63385127355204
// MI455X (gfx1250) — matrix operand order repaired and checked by reading, not compiled, not run on hardware
//
#include <hip/hip_runtime.h>

typedef __attribute__((ext_vector_type(16))) _Float16 v16h;
typedef __attribute__((ext_vector_type(8)))  float    v8f;

#define M_IND  1024
#define DDIM   16
#define JITTERF 1e-3f
#define EPSV    1e-9f

// ---------------------------------------------------------------------------
// 1) Scale inducing points by 1/|ls| and compute row norms
// ---------------------------------------------------------------------------
__global__ void prep_xm_kernel(const float* __restrict__ xm, const float* __restrict__ ls,
                               float* __restrict__ xsm, float* __restrict__ bn) {
    int m = blockIdx.x * blockDim.x + threadIdx.x;
    if (m < M_IND) {
        float s = 0.f;
        #pragma unroll
        for (int d = 0; d < DDIM; ++d) {
            float l = fabsf(ls[d]) + EPSV;
            float v = xm[m * DDIM + d] / l;
            xsm[m * DDIM + d] = v;
            s += v * v;
        }
        bn[m] = s;
    }
}

// ---------------------------------------------------------------------------
// 2) K_mm = sv * exp(-0.5 d2) + jitter * I   (f32, 1024x1024)
// ---------------------------------------------------------------------------
__global__ void kmm_kernel(const float* __restrict__ xsm, const float* __restrict__ bn,
                           const float* __restrict__ sv, float* __restrict__ A) {
    int idx = blockIdx.x * 256 + threadIdx.x;     // 1M elements
    int i = idx >> 10, j = idx & 1023;
    float dot = 0.f;
    #pragma unroll
    for (int d = 0; d < DDIM; ++d) dot += xsm[i * DDIM + d] * xsm[j * DDIM + d];
    float d2 = fmaxf(bn[i] + bn[j] - 2.f * dot, 0.f);
    float v = fabsf(sv[0]) * __expf(-0.5f * d2);
    if (i == j) v += JITTERF;
    A[idx] = v;
}

// ---------------------------------------------------------------------------
// 3) In-place Cholesky (lower), single workgroup of 1024 threads.
//    Right-looking; trailing rank-1 update is load-balanced over the
//    triangular index space.
// ---------------------------------------------------------------------------
__global__ void chol_kernel(float* __restrict__ A) {
    const int t = threadIdx.x;
    for (int k = 0; k < M_IND; ++k) {
        if (t == 0) A[k * M_IND + k] = sqrtf(fmaxf(A[k * M_IND + k], 1e-20f));
        __syncthreads();
        float dk = A[k * M_IND + k];
        int i = k + 1 + t;                 // 1024 threads cover all remaining rows
        if (i < M_IND) A[i * M_IND + k] /= dk;
        __syncthreads();
        int n = M_IND - k - 1;
        long total = (long)n * (n + 1) / 2;
        for (long f = t; f < total; f += 1024) {
            float fr = (sqrtf(8.f * (float)f + 1.f) - 1.f) * 0.5f;
            int r = (int)fr;
            while ((long)r * (r + 1) / 2 > f) --r;
            while ((long)(r + 1) * (r + 2) / 2 <= f) ++r;
            int ii = k + 1 + r;
            int jj = k + 1 + (int)(f - (long)r * (r + 1) / 2);
            A[ii * M_IND + jj] -= A[ii * M_IND + k] * A[jj * M_IND + k];
        }
        __syncthreads();
    }
}

// ---------------------------------------------------------------------------
// 4) T = L^{-1} (lower-triangular inverse).  One wave (32 lanes) per column:
//    forward substitution with shuffle reductions.
// ---------------------------------------------------------------------------
__global__ void trinv_kernel(const float* __restrict__ A, float* __restrict__ T) {
    const int col  = blockIdx.x;
    const int lane = threadIdx.x;      // 0..31 (wave32)
    __shared__ float x[M_IND];
    for (int i = lane; i < col; i += 32) T[i * M_IND + col] = 0.f;
    if (lane == 0) x[col] = 1.0f / A[col * M_IND + col];
    __syncthreads();
    for (int i = col + 1; i < M_IND; ++i) {
        float s = 0.f;
        for (int j = col + lane; j < i; j += 32) s += A[i * M_IND + j] * x[j];
        #pragma unroll
        for (int o = 16; o > 0; o >>= 1) s += __shfl_down(s, o);
        if (lane == 0) x[i] = -s / A[i * M_IND + i];
        __syncthreads();
    }
    for (int i = col + lane; i < M_IND; i += 32) T[i * M_IND + col] = x[i];
}

// ---------------------------------------------------------------------------
// 5) w1 = T^T (T qu),  w2 = T^T (T chole)   (tiny, one block)
// ---------------------------------------------------------------------------
__global__ void wvec_kernel(const float* __restrict__ T, const float* __restrict__ qu,
                            const float* __restrict__ ch,
                            float* __restrict__ w1, float* __restrict__ w2) {
    __shared__ float t1[M_IND], t2[M_IND];
    int t = threadIdx.x;
    for (int i = t; i < M_IND; i += 256) {
        float s1 = 0.f, s2 = 0.f;
        for (int j = 0; j <= i; ++j) {
            float Tv = T[i * M_IND + j];
            s1 += Tv * qu[j];
            s2 += Tv * ch[j];
        }
        t1[i] = s1; t2[i] = s2;
    }
    __syncthreads();
    for (int j = t; j < M_IND; j += 256) {
        float s1 = 0.f, s2 = 0.f;
        for (int i = j; i < M_IND; ++i) {
            float Tv = T[i * M_IND + j];
            s1 += Tv * t1[i];
            s2 += Tv * t2[i];
        }
        w1[j] = s1; w2[j] = s2;
    }
}

// ---------------------------------------------------------------------------
// 6) Pack U = T^T into f16 WMMA B-fragment layout.
//    B (32x16, KxN): lane = ((K>>3)&1)*16 + (N&15);
//    element j = (K&7) + (K&16 ? 8 : 0).  Stored so each lane reads one
//    contiguous 32B v16h: Upk[((c*64 + nb)*32 + lane)*16 + j]
// ---------------------------------------------------------------------------
__global__ void packU_kernel(const float* __restrict__ T, _Float16* __restrict__ Upk) {
    int idx = blockIdx.x * 256 + threadIdx.x;   // over (k, n)
    int k = idx >> 10, n = idx & 1023;
    int c    = k >> 5;
    int kk   = k & 31;
    int lane = ((kk >> 3) & 1) * 16 + (n & 15);
    int j    = (kk & 7) + ((kk & 16) ? 8 : 0);
    int nb   = n >> 4;
    // U[k][n] = T[n][k]
    Upk[((size_t)(c * 64 + nb) * 32 + (size_t)lane) * 16 + j] = (_Float16)T[n * M_IND + k];
}

// ---------------------------------------------------------------------------
// 7) Main fused kernel: per 16 test rows:
//    Phase A: k-tile [16 x 1024] = ARD(Xte_tile, xm); accumulate k.w1, k.w2;
//             store k-tile into LDS pre-swizzled into WMMA A-fragment layout:
//             A (16x32 MxK): lane = h*16 + M with h=(K>>3)&1;
//             element j = (K&7) + (K&16 ? 8 : 0).
//    Phase B: Z = ktile @ U via v_wmma_f32_16x16x32_f16, q_i += sum_n Z_in^2
//             using the 16x16 f32 C/D layout (VGPR r: M=r / r+8 by lane half).
// ---------------------------------------------------------------------------
__global__ void __launch_bounds__(256) svgp_main_kernel(
    const float* __restrict__ Xte, const float* __restrict__ ls, const float* __restrict__ sv,
    const float* __restrict__ xsm, const float* __restrict__ bn,
    const float* __restrict__ w1,  const float* __restrict__ w2,
    const _Float16* __restrict__ Upk, float* __restrict__ out, int N)
{
    __shared__ __align__(32) _Float16 kLds[32 * 32 * 16];   // [chunk c][lane][elem j] 32KB
    __shared__ float aRow[16][17];
    __shared__ float an[16];
    __shared__ float meanAcc[16], vAcc[16], qAcc[16];

    const int t    = threadIdx.x;
    const int tile = blockIdx.x;

    if (t < 16) { meanAcc[t] = 0.f; vAcc[t] = 0.f; qAcc[t] = 0.f; }
    {   // scaled test rows for this tile
        int r = t >> 4, d = t & 15;
        int row = tile * 16 + r;
        float l = fabsf(ls[d]) + EPSV;
        float x = (row < N) ? Xte[row * DDIM + d] : 0.f;
        aRow[r][d] = x / l;
    }
    __syncthreads();
    if (t < 16) {
        float s = 0.f;
        #pragma unroll
        for (int d = 0; d < DDIM; ++d) s += aRow[t][d] * aRow[t][d];
        an[t] = s;
    }
    __syncthreads();

    const float svAbs = fabsf(sv[0]);

    // ---- Phase A: k-tile + small dot products -----------------------------
    {
        const int i = t & 15;      // test row within tile (M index)
        const int g = t >> 4;      // m-group
        float s1 = 0.f, s2 = 0.f;
        for (int mm = 0; mm < 64; ++mm) {
            int m = g * 64 + mm;   // K index, 0..1023
            float dot = 0.f;
            #pragma unroll
            for (int d = 0; d < DDIM; ++d) dot += aRow[i][d] * xsm[m * DDIM + d];
            float d2 = fmaxf(an[i] + bn[m] - 2.f * dot, 0.f);
            float kv = svAbs * __expf(-0.5f * d2);
            s1 += kv * w1[m];
            s2 += kv * w2[m];
            int c    = m >> 5;
            int kk   = m & 31;
            int h    = (kk >> 3) & 1;
            int j    = (kk & 7) + ((kk & 16) ? 8 : 0);
            int lane = h * 16 + i;
            kLds[(c * 32 + lane) * 16 + j] = (_Float16)kv;
        }
        atomicAdd(&meanAcc[i], s1);
        atomicAdd(&vAcc[i], s2);
    }
    __syncthreads();

    // ---- Phase B: q_i = sum_n (ktile @ U)^2 via f16 WMMA -----------------
    {
        const int wv   = t >> 5;   // wave id 0..7
        const int lane = t & 31;
        float q[8];
        #pragma unroll
        for (int r = 0; r < 8; ++r) q[r] = 0.f;

        for (int nb = wv; nb < 64; nb += 8) {        // 16-wide N blocks
            v8f acc = {0.f, 0.f, 0.f, 0.f, 0.f, 0.f, 0.f, 0.f};
            for (int c = 0; c < 32; ++c) {           // K = 1024 in steps of 32
                v16h av = *(const v16h*)(&kLds[(c * 32 + lane) * 16]);
                v16h bv = *(const v16h*)(Upk + ((size_t)(c * 64 + nb) * 32 + (size_t)lane) * 16);
                acc = __builtin_amdgcn_wmma_f32_16x16x32_f16(
                    /*neg_a=*/false, av, /*neg_b=*/false, bv,
                    /*c_mod=*/(short)0, acc, /*reuse_a=*/false, /*reuse_b=*/false);
            }
            #pragma unroll
            for (int r = 0; r < 8; ++r) q[r] += acc[r] * acc[r];
        }
        int rowBase = (lane >= 16) ? 8 : 0;          // C/D layout: lanes 16-31 hold M=r+8
        #pragma unroll
        for (int r = 0; r < 8; ++r) atomicAdd(&qAcc[r + rowBase], q[r]);
    }
    __syncthreads();

    if (t < 16) {
        int row = tile * 16 + t;
        if (row < N) {
            float v = vAcc[t];
            out[row]     = meanAcc[t];                       // mean
            out[N + row] = svAbs - qAcc[t] + v * v;          // yvar
        }
    }
}

// ---------------------------------------------------------------------------
// Launcher
// ---------------------------------------------------------------------------
extern "C" void kernel_launch(void* const* d_in, const int* in_sizes, int n_in,
                              void* d_out, int out_size, void* d_ws, size_t ws_size,
                              hipStream_t stream) {
    const float* Xte = (const float*)d_in[0];
    const float* xm  = (const float*)d_in[1];
    const float* qu  = (const float*)d_in[2];
    const float* ch  = (const float*)d_in[3];
    const float* ls  = (const float*)d_in[4];
    const float* sv  = (const float*)d_in[5];
    float* out = (float*)d_out;

    char* ws = (char*)d_ws;
    float*    A   = (float*)(ws);                          // 4 MB: K_mm -> L
    float*    T   = (float*)(ws + (size_t)(4u << 20));     // 4 MB: L^{-1}
    _Float16* Upk = (_Float16*)(ws + (size_t)(8u << 20));  // 2 MB: packed U f16
    float*    xsm = (float*)(ws + (size_t)(10u << 20));    // 64 KB
    float*    bn  = xsm + M_IND * DDIM;                    // 4 KB
    float*    w1  = bn + M_IND;                            // 4 KB
    float*    w2  = w1 + M_IND;                            // 4 KB

    int N = in_sizes[0] / DDIM;                            // 20000

    prep_xm_kernel<<<(M_IND + 255) / 256, 256, 0, stream>>>(xm, ls, xsm, bn);
    kmm_kernel<<<(M_IND * M_IND) / 256, 256, 0, stream>>>(xsm, bn, sv, A);
    chol_kernel<<<1, 1024, 0, stream>>>(A);
    trinv_kernel<<<M_IND, 32, 0, stream>>>(A, T);
    wvec_kernel<<<1, 256, 0, stream>>>(T, qu, ch, w1, w2);
    packU_kernel<<<(M_IND * M_IND) / 256, 256, 0, stream>>>(T, Upk);
    int tiles = (N + 15) / 16;
    svgp_main_kernel<<<tiles, 256, 0, stream>>>(Xte, ls, sv, xsm, bn, w1, w2, Upk, out, N);
}
